// _EncoderBlock_21964462752254
// MI455X (gfx1250) — matrix operand order repaired and checked by reading, not compiled, not run on hardware
//
#include <hip/hip_runtime.h>

typedef __attribute__((ext_vector_type(16))) _Float16 v16h;
typedef __attribute__((ext_vector_type(8)))  _Float16 v8h;
typedef __attribute__((ext_vector_type(8)))  float    v8f;

#define D8    8
#define C_CH  256
#define FEAT  2048   // C*D
#define TOK   4096   // B*N
#define NROT  4

// Cayley sign for Cl(3,0,0), Euclidean metric (all +1)
__device__ __forceinline__ float cl_sign(int a, int b) {
    int t = a >> 1, cnt = 0;
    while (t) { cnt += __popc(t & b); t >>= 1; }
    return (cnt & 1) ? -1.0f : 1.0f;
}

// ---- CDNA5 async memory->LDS copy (ASYNCcnt path), 32B per call ----
// IOFFSET is added to BOTH the LDS and global address (ISA 15.18 async pseudocode),
// so one base + offset:16 moves 32 contiguous bytes.
__device__ __forceinline__ void async_copy_2x128(const _Float16* g, unsigned lds_byte_off) {
    unsigned long long ga = (unsigned long long)(size_t)g;
    asm volatile("global_load_async_to_lds_b128 %0, %1, off\n\t"
                 "global_load_async_to_lds_b128 %0, %1, off offset:16"
                 :: "v"(lds_byte_off), "v"(ga)
                 : "memory");
}

#if __has_builtin(__builtin_amdgcn_s_wait_asynccnt)
#define WAIT_ASYNC(n) __builtin_amdgcn_s_wait_asynccnt(n)
#else
#define WAIT_ASYNC(n) asm volatile("s_wait_asynccnt %0" :: "n"(n) : "memory")
#endif

// ---------------- Kernel 1: mixed rotor sandwich matrix M[c] (8x8) ----------------
__global__ void rotor_matrix_kernel(const float* __restrict__ rotor_biv,  // [4][256][3]
                                    const float* __restrict__ rotor_mix,  // [4]
                                    float* __restrict__ Mc_out) {         // [256][8][8]
    int c = blockIdx.x * blockDim.x + threadIdx.x;
    if (c >= C_CH) return;

    float m[NROT], mx = -1e30f;
    for (int r = 0; r < NROT; ++r) { m[r] = rotor_mix[r]; mx = fmaxf(mx, m[r]); }
    float s = 0.f;
    for (int r = 0; r < NROT; ++r) { m[r] = __expf(m[r] - mx); s += m[r]; }

    float Mc[8][8];
    for (int k = 0; k < 8; ++k) for (int j = 0; j < 8; ++j) Mc[k][j] = 0.f;

    const int nz[4] = {0, 3, 5, 6};                       // scalar + bivector blades
    const float rev[8] = {1,1,1,-1,1,-1,-1,-1};           // reversion sign per blade
    for (int r = 0; r < NROT; ++r) {
        float p = m[r] / s;
        float Rv[8] = {0,0,0,0,0,0,0,0};
        Rv[0] = 1.f;
        Rv[3] = rotor_biv[(r*C_CH + c)*3 + 0];
        Rv[5] = rotor_biv[(r*C_CH + c)*3 + 1];
        Rv[6] = rotor_biv[(r*C_CH + c)*3 + 2];
        float inv = 1.f / sqrtf(Rv[0]*Rv[0] + Rv[3]*Rv[3] + Rv[5]*Rv[5] + Rv[6]*Rv[6]);
        for (int d = 0; d < 8; ++d) Rv[d] *= inv;
        float Rt[8];
        for (int d = 0; d < 8; ++d) Rt[d] = Rv[d] * rev[d];

        float T1[8][8], Y[8][8];
        for (int a = 0; a < 8; ++a) for (int b = 0; b < 8; ++b) { T1[a][b] = 0.f; Y[a][b] = 0.f; }
        // t_k = sum_{i,j} R_i h_j C[i,j,k]  ->  T1[k][j]
        for (int u = 0; u < 4; ++u) {
            int i = nz[u];
            for (int j = 0; j < 8; ++j) T1[i ^ j][j] += Rv[i] * cl_sign(i, j);
        }
        // y_k = sum_{i,j2} t_i Rt_j2 C[i,j2,k]  ->  Y[k][i]
        for (int u = 0; u < 4; ++u) {
            int j2 = nz[u];
            for (int i = 0; i < 8; ++i) Y[i ^ j2][i] += Rt[j2] * cl_sign(i, j2);
        }
        for (int k = 0; k < 8; ++k)
            for (int j = 0; j < 8; ++j) {
                float acc = 0.f;
                for (int t = 0; t < 8; ++t) acc += Y[k][t] * T1[t][j];
                Mc[k][j] += p * acc;
            }
    }
    float* o = Mc_out + (size_t)c * 64;
    for (int k = 0; k < 8; ++k)
        for (int j = 0; j < 8; ++j) o[k*8 + j] = Mc[k][j];
}

// ---------------- Kernel 2: fused weight  WfT[n=(o,k)][f=(c,j)]  (f16) ----------------
__global__ void fuse_weight_kernel(const float* __restrict__ lin_weight, // [256][256][8]
                                   const float* __restrict__ Mc_in,      // [256][8][8]
                                   _Float16* __restrict__ WfT) {         // [2048][2048]
    int tid = blockIdx.x * blockDim.x + threadIdx.x;   // o*256 + c
    if (tid >= C_CH * C_CH) return;
    int o = tid >> 8, c = tid & 255;

    float w8[8];
    for (int i = 0; i < 8; ++i) w8[i] = lin_weight[(size_t)tid * 8 + i];

    float Wb[8][8];
    for (int k = 0; k < 8; ++k) for (int j = 0; j < 8; ++j) Wb[k][j] = 0.f;
    // Wb[k][j'] = sum_i w_i C[i,j',k]
    for (int i = 0; i < 8; ++i)
        for (int jp = 0; jp < 8; ++jp) Wb[i ^ jp][jp] += w8[i] * cl_sign(i, jp);

    float Mc[8][8];
    for (int a = 0; a < 8; ++a)
        for (int b = 0; b < 8; ++b) Mc[a][b] = Mc_in[(size_t)c*64 + a*8 + b];

    for (int k = 0; k < 8; ++k)
        for (int j = 0; j < 8; ++j) {
            float acc = 0.f;
            for (int jp = 0; jp < 8; ++jp) acc += Wb[k][jp] * Mc[jp][j];
            WfT[(size_t)(o*8 + k) * FEAT + (c*8 + j)] = (_Float16)acc;
        }
}

// ---------------- Kernel 3: grade-norm + scale + GELU, f32 -> f16 ----------------
__global__ void act_kernel(const float* __restrict__ x,            // [4096][256][8]
                           const float* __restrict__ norm_weight,  // [256][4]
                           _Float16* __restrict__ act) {           // [4096][2048]
    int tid = blockIdx.x * blockDim.x + threadIdx.x;  // t*256 + c
    if (tid >= TOK * C_CH) return;
    int c = tid & 255;

    const float4* xp = (const float4*)(x + (size_t)tid * 8);
    float4 a = xp[0], b = xp[1];
    float v[8] = {a.x, a.y, a.z, a.w, b.x, b.y, b.z, b.w};

    const float EPS = 1e-5f;
    float g[4];
    g[0] = sqrtf(v[0]*v[0] + EPS);
    g[1] = sqrtf(v[1]*v[1] + v[2]*v[2] + v[4]*v[4] + EPS);
    g[2] = sqrtf(v[3]*v[3] + v[5]*v[5] + v[6]*v[6] + EPS);
    g[3] = sqrtf(v[7]*v[7] + EPS);
    float w[4];
    for (int q = 0; q < 4; ++q) w[q] = norm_weight[c*4 + q] / g[q];

    const int grade[8] = {0,1,1,2,1,2,2,3};
    _Float16 h8[8];
    for (int d = 0; d < 8; ++d) {
        float h = v[d] * w[grade[d]];
        float t = tanhf(0.7978845608028654f * (h + 0.044715f * h*h*h));
        h8[d] = (_Float16)(0.5f * h * (1.f + t));
    }
    *(float4*)(act + (size_t)tid * 8) = *(float4*)h8;   // 8 halves = 16B
}

// ---------------- Kernel 4: WMMA GEMM (async-LDS double-buffered) + bias + residual ----
#define BM 128
#define BN 128
#define BK 32
#define NK (FEAT / BK)   // 64
#define LDA 40           // padded row stride (halves) to dodge LDS bank conflicts

__global__ __launch_bounds__(256)
void gemm_kernel(const _Float16* __restrict__ act,  // A [4096][2048]
                 const _Float16* __restrict__ WfT,  // B^T [2048 n][2048 k]
                 const float* __restrict__ bias,    // [2048]
                 const float* __restrict__ xres,    // residual [4096][2048]
                 float* __restrict__ out) {         // [4096][2048]
    __shared__ _Float16 As[2][BM * LDA];
    __shared__ _Float16 Bs[2][BN * LDA];

    int tid  = threadIdx.x;
    int wave = tid >> 5, lane = tid & 31;
    int wm = wave >> 1, wn = wave & 1;          // 4x2 wave grid -> 32x64 per wave
    int m0 = blockIdx.y * BM, n0 = blockIdx.x * BN;

    v8f zero = {};
    v8f acc[2][4];
    for (int i = 0; i < 2; ++i) for (int j = 0; j < 4; ++j) acc[i][j] = zero;

    int lrow = tid >> 1;             // 0..127 : tile row this thread stages
    int lcol = (tid & 1) * 16;       // 0 or 16 halves within the 32-wide K slab

    const _Float16* ga = act + (size_t)(m0 + lrow) * FEAT + lcol;
    const _Float16* gb = WfT + (size_t)(n0 + lrow) * FEAT + lcol;
    // flat->LDS mapping is low-32-bit truncation (ISA 10.2), so this is the LDS byte offset
    unsigned la0 = (unsigned)(size_t)&As[0][lrow * LDA + lcol];
    unsigned lb0 = (unsigned)(size_t)&Bs[0][lrow * LDA + lcol];
    unsigned la1 = (unsigned)(size_t)&As[1][lrow * LDA + lcol];
    unsigned lb1 = (unsigned)(size_t)&Bs[1][lrow * LDA + lcol];

    // prologue: stage tile 0 into buffer 0 (4 async b128 per thread)
    async_copy_2x128(ga, la0);
    async_copy_2x128(gb, lb0);

    for (int kb = 0; kb < NK; ++kb) {
        int cur = kb & 1;
        if (kb + 1 < NK) {
            // stage next tile into the other buffer while we compute
            async_copy_2x128(ga + (kb + 1) * BK, cur ? la0 : la1);
            async_copy_2x128(gb + (kb + 1) * BK, cur ? lb0 : lb1);
            WAIT_ASYNC(4);   // async loads finish in order: tile kb's 4 are done
        } else {
            WAIT_ASYNC(0);
        }
        __syncthreads();     // all waves see tile kb in LDS

        const _Float16* Ab = &As[cur][0];
        const _Float16* Bb = &Bs[cur][0];
        int rl  = lane & 15;
        int kb8 = (lane >> 4) << 3;   // A: K base 0/8, pair chunk at +16
        int ks  = (lane >> 4) << 4;   // unused: B is read with kb8 like A (K base 0/8, pair chunk at +16)
        union Frag { v16h v; v8h h[2]; } af[2], bf[4];
        for (int ms = 0; ms < 2; ++ms) {
            int r = wm * 32 + ms * 16 + rl;
            af[ms].h[0] = *(const v8h*)&Ab[r * LDA + kb8];
            af[ms].h[1] = *(const v8h*)&Ab[r * LDA + kb8 + 16];
        }
        for (int ns = 0; ns < 4; ++ns) {
            int r = wn * 64 + ns * 16 + rl;
            bf[ns].h[0] = *(const v8h*)&Bb[r * LDA + kb8];
            bf[ns].h[1] = *(const v8h*)&Bb[r * LDA + kb8 + 16];
        }
        for (int ms = 0; ms < 2; ++ms)
            for (int ns = 0; ns < 4; ++ns)
                acc[ms][ns] = __builtin_amdgcn_wmma_f32_16x16x32_f16(
                    false, af[ms].v, false, bf[ns].v,
                    (short)0, acc[ms][ns], false, false);

        __syncthreads();     // everyone done reading buf[cur] before it is re-filled
    }

    // epilogue: D layout -> lane: N = lane%16, M = r + 8*(lane>=16)
    for (int ms = 0; ms < 2; ++ms)
        for (int ns = 0; ns < 4; ++ns) {
            int colg = n0 + wn * 64 + ns * 16 + (lane & 15);
            int rb   = m0 + wm * 32 + ms * 16 + ((lane >> 4) << 3);
            float bcol = bias[colg];
            for (int r = 0; r < 8; ++r) {
                size_t idx = (size_t)(rb + r) * FEAT + colg;
                out[idx] = acc[ms][ns][r] + bcol + xres[idx];
            }
        }
}

extern "C" void kernel_launch(void* const* d_in, const int* in_sizes, int n_in,
                              void* d_out, int out_size, void* d_ws, size_t ws_size,
                              hipStream_t stream) {
    const float* x           = (const float*)d_in[0];
    const float* norm_weight = (const float*)d_in[1];
    const float* rotor_biv   = (const float*)d_in[2];
    const float* rotor_mix   = (const float*)d_in[3];
    const float* lin_weight  = (const float*)d_in[4];
    const float* lin_bias    = (const float*)d_in[5];
    float* out = (float*)d_out;

    // workspace layout: Mc (64KB) | WfT f16 (8MB) | act f16 (16MB)
    float*    Mc  = (float*)d_ws;
    _Float16* WfT = (_Float16*)((char*)d_ws + 65536);
    _Float16* act = (_Float16*)((char*)d_ws + 65536 + (size_t)FEAT * FEAT * 2);

    rotor_matrix_kernel<<<1, 256, 0, stream>>>(rotor_biv, rotor_mix, Mc);
    fuse_weight_kernel<<<(C_CH * C_CH) / 256, 256, 0, stream>>>(lin_weight, Mc, WfT);
    act_kernel<<<(TOK * C_CH) / 256, 256, 0, stream>>>(x, norm_weight, act);
    gemm_kernel<<<dim3(FEAT / BN, TOK / BM), 256, 0, stream>>>(act, WfT, lin_bias, x, out);
}
